// ThreeBodyGhostNet_25984552141464
// MI455X (gfx1250) — matrix operand order repaired and checked by reading, not compiled, not run on hardware
//
#include <hip/hip_runtime.h>

#define N_NODES 50000
#define N_EDGES 640000
#define HIDDEN  128
#define TILE_E  32            // edges per block iteration (2 M-tiles of 16)
#define LAYER_W 49152         // bf16 elems per layer: 128*256 (W1^T) + 128*128 (W2^T)

typedef __attribute__((ext_vector_type(16))) __bf16 v16bf;
typedef __attribute__((ext_vector_type(8)))  float  v8f;

union Frag {
    v16bf v;
    uint4 q[2];
};

__device__ __forceinline__ unsigned short f2bf(float f) {
    unsigned int u = __float_as_uint(f);
    u += 0x7FFFu + ((u >> 16) & 1u);          // round-to-nearest-even
    return (unsigned short)(u >> 16);
}
__device__ __forceinline__ float bf2f(unsigned short u) {
    return __uint_as_float(((unsigned int)u) << 16);
}

// ---------------------------------------------------------------------------
// One-shot: transpose + convert all layers' W1/W2 to bf16 in workspace.
// Layer l: wbuf + l*LAYER_W, [0,32768) = W1^T [n][k=256], [32768,49152) = W2^T [n][k=128]
// ---------------------------------------------------------------------------
__global__ __launch_bounds__(256) void prep_weights_kernel(
    const float* __restrict__ W1, const float* __restrict__ W2,
    unsigned short* __restrict__ wbuf)
{
    const int l = blockIdx.y;
    const float* w1 = W1 + (size_t)l * 257 * 128;
    const float* w2 = W2 + (size_t)l * 128 * 128;
    unsigned short* o = wbuf + (size_t)l * LAYER_W;
    for (int i = blockIdx.x * 256 + threadIdx.x; i < LAYER_W; i += gridDim.x * 256) {
        if (i < 32768) {
            int n = i >> 8, k = i & 255;
            o[i] = f2bf(w1[(size_t)k * 128 + n]);
        } else {
            int j = i - 32768;
            int n = j >> 7, k = j & 127;
            o[i] = f2bf(w2[(size_t)k * 128 + n]);
        }
    }
}

// ---------------------------------------------------------------------------
// Encoder: h = x @ W_enc + b_enc   ([N,4] x [4,128]) -> bf16 node features
// ---------------------------------------------------------------------------
__global__ __launch_bounds__(256) void encoder_kernel(
    const float* __restrict__ x, const float* __restrict__ W,
    const float* __restrict__ b, unsigned short* __restrict__ hb)
{
    __shared__ float sW[4 * HIDDEN];
    __shared__ float sB[HIDDEN];
    for (int i = threadIdx.x; i < 4 * HIDDEN; i += 256) sW[i] = W[i];
    if (threadIdx.x < HIDDEN) sB[threadIdx.x] = b[threadIdx.x];
    __syncthreads();

    int gid = blockIdx.x * 256 + threadIdx.x;
    if (gid < N_NODES * HIDDEN) {
        int n = gid >> 7, c = gid & 127;
        const float4 xv = *(const float4*)(x + (size_t)n * 4);
        float v = xv.x * sW[c] + xv.y * sW[HIDDEN + c] +
                  xv.z * sW[2 * HIDDEN + c] + xv.w * sW[3 * HIDDEN + c] + sB[c];
        hb[gid] = f2bf(v);
    }
}

__global__ __launch_bounds__(256) void zero_kernel(float* __restrict__ p) {
    size_t i = (size_t)blockIdx.x * 256 + threadIdx.x;
    if (i < (size_t)N_NODES * HIDDEN) p[i] = 0.0f;
}

// h_bf16 = relu(agg); optionally mirror f32 result into d_out (final layer)
__global__ __launch_bounds__(256) void relu_copy_kernel(
    const float* __restrict__ agg, unsigned short* __restrict__ hb,
    float* __restrict__ out)
{
    size_t i = (size_t)blockIdx.x * 256 + threadIdx.x;
    if (i < (size_t)N_NODES * HIDDEN) {
        float v = agg[i];
        v = v > 0.0f ? v : 0.0f;
        hb[i] = f2bf(v);
        if (out) out[i] = v;
    }
}

// ---------------------------------------------------------------------------
// Edge message-passing layer, double-buffered:
//   while tile t's 24 WMMAs run, tile t+1's random node rows are prefetched
//   from the L2-resident bf16 table into VGPRs; ds_store happens next iter.
// ---------------------------------------------------------------------------
__global__ __launch_bounds__(256) void edge_layer_kernel(
    const unsigned short* __restrict__ hb,
    const int*   __restrict__ src,
    const int*   __restrict__ dst,
    const float* __restrict__ ea,
    const unsigned short* __restrict__ wb16,   // this layer's bf16 W1^T|W2^T
    const float* __restrict__ W1,              // f32, for edge_attr row
    const float* __restrict__ b1,
    const float* __restrict__ b2,
    float* __restrict__ agg)
{
    __shared__ unsigned short sW1[128 * 256];        // 64 KB : W1^T bf16 [n][k]
    __shared__ unsigned short sW2[128 * 128];        // 32 KB : W2^T bf16 [n][k]
    __shared__ unsigned short sA [2][TILE_E * 256];  // 2x16 KB : concat tiles
    __shared__ unsigned short sA2[TILE_E * 128];     //  8 KB : hidden tile
    __shared__ float sW1last[128], sB1[128], sB2[128];
    __shared__ float sEA[2][TILE_E];
    __shared__ int   sDst[2][TILE_E];

    const int tid   = threadIdx.x;
    const int lane  = tid & 31;
    const int wv    = tid >> 5;
    const int mrow  = lane & 15;        // A-row / B-col owned by this lane
    const int khalf = lane >> 4;        // K half-group
    const int gm    = tid >> 3;         // gather: edge row owned by this thread
    const int gseg  = tid & 7;          // gather: 32-bf16 segment (64 B)

    // ---- stage pre-converted bf16 weights (pure b128 copies) ----
    {
        const uint4* wsrc = (const uint4*)wb16;      // 6144 uint4 total
        uint4* d1 = (uint4*)sW1;                     // 4096 uint4
        uint4* d2 = (uint4*)sW2;                     // 2048 uint4
        for (int i = tid; i < 4096; i += 256) d1[i] = wsrc[i];
        for (int i = tid; i < 2048; i += 256) d2[i] = wsrc[4096 + i];
    }
    if (tid < 128) {
        sW1last[tid] = W1[256 * 128 + tid];   // edge_attr row of W1 (f32)
        sB1[tid]     = b1[tid];
        sB2[tid]     = b2[tid];
    }

    const int nTiles = N_EDGES / TILE_E;
    const int stride = gridDim.x;
    int t   = blockIdx.x;
    int buf = 0;
    uint4 g[4];                          // prefetched 64 B of node features

    // ---- prologue: prefetch first tile ----
    if (t < nTiles) {
        if (tid < TILE_E) {
            sDst[0][tid] = dst[t * TILE_E + tid];
            sEA [0][tid] = ea [t * TILE_E + tid];
        }
        int e = t * TILE_E + gm;
        int node = (gseg < 4) ? dst[e] : src[e];
        const uint4* inp = (const uint4*)(hb + (size_t)node * HIDDEN + (gseg & 3) * 32);
        g[0] = inp[0]; g[1] = inp[1]; g[2] = inp[2]; g[3] = inp[3];
    }

    for (; t < nTiles; t += stride, buf ^= 1) {
        // ---- commit prefetched rows into this tile's A buffer ----
        {
            uint4* outp = (uint4*)&sA[buf][gm * 256 + gseg * 32];
            outp[0] = g[0]; outp[1] = g[1]; outp[2] = g[2]; outp[3] = g[3];
        }
        __syncthreads();   // sA[buf], sDst[buf], sEA[buf], weights ready

        // ---- prefetch next tile (overlaps with all WMMA work below) ----
        {
            int tn = t + stride;
            int tc = (tn < nTiles) ? tn : t;   // clamp; redundant on last iter
            if (tid < TILE_E) {
                sDst[buf ^ 1][tid] = dst[tc * TILE_E + tid];
                sEA [buf ^ 1][tid] = ea [tc * TILE_E + tid];
            }
            int e = tc * TILE_E + gm;
            int node = (gseg < 4) ? dst[e] : src[e];
            const uint4* inp = (const uint4*)(hb + (size_t)node * HIDDEN + (gseg & 3) * 32);
            g[0] = inp[0]; g[1] = inp[1]; g[2] = inp[2]; g[3] = inp[3];
        }

        // ---- GEMM1: K = 256, two M-tiles share each B fragment ----
        v8f acc0 = {}, acc1 = {};
        #pragma unroll
        for (int ks = 0; ks < 8; ++ks) {
            Frag a0, a1, b;
            const unsigned short* wb = &sW1[(wv * 16 + mrow) * 256 + ks * 32 + khalf * 8];
            b.q[0] = ((const uint4*)wb)[0];
            b.q[1] = ((const uint4*)wb)[2];
            a0.q[0] = *(const uint4*)&sA[buf][mrow * 256 + ks * 32 + khalf * 8];
            a0.q[1] = *(const uint4*)&sA[buf][mrow * 256 + ks * 32 + khalf * 8 + 16];
            a1.q[0] = *(const uint4*)&sA[buf][(16 + mrow) * 256 + ks * 32 + khalf * 8];
            a1.q[1] = *(const uint4*)&sA[buf][(16 + mrow) * 256 + ks * 32 + khalf * 8 + 16];
            acc0 = __builtin_amdgcn_wmma_f32_16x16x32_bf16(
                false, a0.v, false, b.v, (short)0, acc0, false, false);
            acc1 = __builtin_amdgcn_wmma_f32_16x16x32_bf16(
                false, a1.v, false, b.v, (short)0, acc1, false, false);
        }

        // ---- epilogue 1: rank-1 edge_attr term + bias + ReLU -> bf16 A2 ----
        {
            const int c   = wv * 16 + (lane & 15);
            const int mhi = (lane >> 4) * 8;
            const float wl = sW1last[c], bb = sB1[c];
            #pragma unroll
            for (int v = 0; v < 8; ++v) {
                int m0 = mhi + v;
                float x0 = acc0[v] + sEA[buf][m0] * wl + bb;
                float x1 = acc1[v] + sEA[buf][16 + m0] * wl + bb;
                x0 = x0 > 0.0f ? x0 : 0.0f;
                x1 = x1 > 0.0f ? x1 : 0.0f;
                sA2[m0 * 128 + c]        = f2bf(x0);
                sA2[(16 + m0) * 128 + c] = f2bf(x1);
            }
        }
        __syncthreads();

        // ---- GEMM2: K = 128, two M-tiles share each B fragment ----
        v8f acc2 = {}, acc3 = {};
        #pragma unroll
        for (int ks = 0; ks < 4; ++ks) {
            Frag a0, a1, b;
            const unsigned short* wb = &sW2[(wv * 16 + mrow) * 128 + ks * 32 + khalf * 8];
            b.q[0] = ((const uint4*)wb)[0];
            b.q[1] = ((const uint4*)wb)[2];
            a0.q[0] = *(const uint4*)&sA2[mrow * 128 + ks * 32 + khalf * 8];
            a0.q[1] = *(const uint4*)&sA2[mrow * 128 + ks * 32 + khalf * 8 + 16];
            a1.q[0] = *(const uint4*)&sA2[(16 + mrow) * 128 + ks * 32 + khalf * 8];
            a1.q[1] = *(const uint4*)&sA2[(16 + mrow) * 128 + ks * 32 + khalf * 8 + 16];
            acc2 = __builtin_amdgcn_wmma_f32_16x16x32_bf16(
                false, a0.v, false, b.v, (short)0, acc2, false, false);
            acc3 = __builtin_amdgcn_wmma_f32_16x16x32_bf16(
                false, a1.v, false, b.v, (short)0, acc3, false, false);
        }

        // ---- epilogue 2: bias + scatter-add (L2-resident atomics) ----
        {
            const int c   = wv * 16 + (lane & 15);
            const int mhi = (lane >> 4) * 8;
            const float bb = sB2[c];
            #pragma unroll
            for (int v = 0; v < 8; ++v) {
                int m0 = mhi + v;
                atomicAdd(&agg[(size_t)sDst[buf][m0] * HIDDEN + c],      acc2[v] + bb);
                atomicAdd(&agg[(size_t)sDst[buf][16 + m0] * HIDDEN + c], acc3[v] + bb);
            }
        }
    }
}

// ---------------------------------------------------------------------------
// Heads: one wave per node, lane k owns features [4k, 4k+4)
// ---------------------------------------------------------------------------
__global__ __launch_bounds__(256) void heads_kernel(
    const unsigned short* __restrict__ hb,
    const float* __restrict__ Wg,  const float* __restrict__ bg,
    const float* __restrict__ Wp,  const float* __restrict__ bp,
    const float* __restrict__ Wgr, const float* __restrict__ bgr,
    float* __restrict__ out)
{
    const int node = blockIdx.x * 8 + (threadIdx.x >> 5);
    const int lane = threadIdx.x & 31;

    const uint2 hq = *(const uint2*)(hb + (size_t)node * HIDDEN + lane * 4);
    float hval[4];
    hval[0] = bf2f((unsigned short)(hq.x & 0xFFFFu));
    hval[1] = bf2f((unsigned short)(hq.x >> 16));
    hval[2] = bf2f((unsigned short)(hq.y & 0xFFFFu));
    hval[3] = bf2f((unsigned short)(hq.y >> 16));

    float g = 0.f, p = 0.f, g0 = 0.f, g1 = 0.f;
    #pragma unroll
    for (int j = 0; j < 4; ++j) {
        int k = lane * 4 + j;
        g  += hval[j] * Wg[k];
        p  += hval[j] * Wp[k];
        g0 += hval[j] * Wgr[k * 2 + 0];
        g1 += hval[j] * Wgr[k * 2 + 1];
    }
    #pragma unroll
    for (int off = 16; off > 0; off >>= 1) {
        g  += __shfl_xor(g,  off, 32);
        p  += __shfl_xor(p,  off, 32);
        g0 += __shfl_xor(g0, off, 32);
        g1 += __shfl_xor(g1, off, 32);
    }
    if (lane == 0) {
        const size_t NH = (size_t)N_NODES * HIDDEN;   // 6,400,000
        out[NH + node]                       = 1.0f / (1.0f + expf(-(g + bg[0])));
        out[NH + N_NODES + node]             = p + bp[0];
        out[NH + 2 * N_NODES + node * 2 + 0] = g0 + bgr[0];
        out[NH + 2 * N_NODES + node * 2 + 1] = g1 + bgr[1];
    }
}

// ---------------------------------------------------------------------------
extern "C" void kernel_launch(void* const* d_in, const int* in_sizes, int n_in,
                              void* d_out, int out_size, void* d_ws, size_t ws_size,
                              hipStream_t stream) {
    const float* x      = (const float*)d_in[0];
    const int*   ei     = (const int*)  d_in[1];   // [2, E]
    const float* ea     = (const float*)d_in[2];
    const float* W_enc  = (const float*)d_in[3];
    const float* b_enc  = (const float*)d_in[4];
    const float* W1     = (const float*)d_in[5];   // [4, 257, 128]
    const float* b1     = (const float*)d_in[6];   // [4, 128]
    const float* W2     = (const float*)d_in[7];   // [4, 128, 128]
    const float* b2     = (const float*)d_in[8];   // [4, 128]
    const float* W_g    = (const float*)d_in[9];
    const float* b_g    = (const float*)d_in[10];
    const float* W_p    = (const float*)d_in[11];
    const float* b_p    = (const float*)d_in[12];
    const float* W_gr   = (const float*)d_in[13];
    const float* b_gr   = (const float*)d_in[14];

    float* out = (float*)d_out;

    // workspace: bf16 h (12.8 MB) | f32 agg (25.6 MB) | bf16 weights (0.4 MB)
    unsigned short* hbuf = (unsigned short*)d_ws;
    size_t off = ((size_t)N_NODES * HIDDEN * sizeof(unsigned short) + 255) & ~(size_t)255;
    float* aggbuf = (float*)((char*)d_ws + off);
    unsigned short* wbuf = (unsigned short*)((char*)aggbuf + (size_t)N_NODES * HIDDEN * sizeof(float));

    const int* src = ei;              // edge_index[0]
    const int* dst = ei + N_EDGES;    // edge_index[1]

    const int elemBlocks = (N_NODES * HIDDEN + 255) / 256;   // 25000

    prep_weights_kernel<<<dim3(64, 4), 256, 0, stream>>>(W1, W2, wbuf);
    encoder_kernel<<<elemBlocks, 256, 0, stream>>>(x, W_enc, b_enc, hbuf);

    for (int l = 0; l < 4; ++l) {
        zero_kernel<<<elemBlocks, 256, 0, stream>>>(aggbuf);
        edge_layer_kernel<<<1280, 256, 0, stream>>>(
            hbuf, src, dst, ea,
            wbuf + (size_t)l * LAYER_W,
            W1 + (size_t)l * 257 * 128, b1 + (size_t)l * 128, b2 + (size_t)l * 128,
            aggbuf);
        relu_copy_kernel<<<elemBlocks, 256, 0, stream>>>(
            aggbuf, hbuf, (l == 3) ? out : (float*)nullptr);
    }

    heads_kernel<<<N_NODES / 8, 256, 0, stream>>>(
        hbuf, W_g, b_g, W_p, b_p, W_gr, b_gr, out);
}
